// TransformerLayer_88399016887077
// MI455X (gfx1250) — matrix operand order repaired and checked by reading, not compiled, not run on hardware
//
#include <hip/hip_runtime.h>
#include <stdint.h>

#define BB 4
#define SS 2048
#define DD 1024
#define MT 64               // query rows per workgroup
#define KR 1024             // keys per round (2 rounds)
#define PST 1032            // padded LDS row stride (bank-conflict-free, 16B aligned)
#define NEGV  (-9.0e15f)
#define SCALE (0.03125f)    // 1/sqrt(1024)

typedef __attribute__((ext_vector_type(16))) __bf16 v16bf;
typedef __attribute__((ext_vector_type(8)))  float  v8f;

__device__ __forceinline__ unsigned short f2bf(float f) {
    unsigned int x = __float_as_uint(f);
    x += 0x7FFFu + ((x >> 16) & 1u);   // round-to-nearest-even
    return (unsigned short)(x >> 16);
}
__device__ __forceinline__ float bf2f(unsigned short u) {
    return __uint_as_float((unsigned int)u << 16);
}

union V16Cast { uint4 q[2]; v16bf v; };

__device__ __forceinline__ v16bf load16bf(const void* lo, const void* hi) {
    V16Cast c;
    c.q[0] = *(const uint4*)lo;
    c.q[1] = *(const uint4*)hi;
    return c.v;
}

// ---------------- kernel 0a: K fp32 -> bf16 (same layout) ----------------
__global__ void __launch_bounds__(256)
cvt_bf16_kernel(const float* __restrict__ in, unsigned short* __restrict__ out) {
    size_t i = ((size_t)blockIdx.x * 256 + threadIdx.x) * 4;
    float4 f = *(const float4*)(in + i);
    unsigned int lo = (unsigned int)f2bf(f.x) | ((unsigned int)f2bf(f.y) << 16);
    unsigned int hi = (unsigned int)f2bf(f.z) | ((unsigned int)f2bf(f.w) << 16);
    *(uint2*)(out + i) = make_uint2(lo, hi);
}

// ---------------- kernel 0b: V fp32 [B][S][D] -> bf16 V^T [B][D][S] ------
__global__ void __launch_bounds__(256)
transpose_v_kernel(const float* __restrict__ v, unsigned short* __restrict__ out) {
    __shared__ unsigned short tile[32][33];
    const int b  = blockIdx.z;
    const int d0 = blockIdx.x * 32;
    const int s0 = blockIdx.y * 32;
#pragma unroll
    for (int j = 0; j < 4; ++j) {
        int srow = s0 + threadIdx.y + j * 8;
        tile[threadIdx.y + j * 8][threadIdx.x] =
            f2bf(v[((size_t)b * SS + srow) * DD + d0 + threadIdx.x]);
    }
    __syncthreads();
#pragma unroll
    for (int j = 0; j < 4; ++j) {
        int d = d0 + threadIdx.y + j * 8;
        out[((size_t)b * DD + d) * SS + s0 + threadIdx.x] = tile[threadIdx.x][threadIdx.y + j * 8];
    }
}

// ---------------- kernel 1: fused flash attention (bf16 WMMA) ------------
// grid: B * S/MT blocks, 512 threads (16 wave32), 1 WG per WGP
__global__ void __launch_bounds__(512)
attn_kernel(const float* __restrict__ q, const unsigned short* __restrict__ kbf,
            const unsigned short* __restrict__ vtb, const int* __restrict__ mask,
            float* __restrict__ xout)
{
    extern __shared__ __align__(16) char smem[];
    unsigned short* Qs   = (unsigned short*)smem;            // MT*PST
    unsigned short* Pall = Qs + MT * PST;                    // MT*PST
    float* red  = (float*)(Pall + MT * PST);                 // 512
    float* mfin = red + 512;                                 // MT
    float* facs = mfin + MT;                                 // MT
    float* lfin = facs + MT;                                 // MT

    const int tid  = threadIdx.x;
    const int w    = tid >> 5;        // 0..15
    const int lane = tid & 31;
    const int b    = blockIdx.x >> 5;
    const int qt   = blockIdx.x & 31;

    // ---- load Q tile (MT rows) to LDS as bf16 with padded stride ----
    const float* qbase = q + ((size_t)b * SS + qt * MT) * DD;
    for (int i = tid; i < MT * DD / 2; i += 512) {
        int row = i >> 9;            // DD/2 = 512 pairs per row
        int cp  = i & 511;
        float2 f = *(const float2*)(qbase + (size_t)row * DD + cp * 2);
        ((unsigned int*)Qs)[row * (PST / 2) + cp] =
            (unsigned int)f2bf(f.x) | ((unsigned int)f2bf(f.y) << 16);
    }
    __syncthreads();

    const int n    = lane & 15;       // A-row (M within subtile), B-col (N), key offset
    const int hsel = lane >> 4;
    const int aoff = hsel * 8;        // A operand K offset / C-D row offset
    const int boff = hsel * 8;        // B operand K offset

    // loop-invariant LDS row bases (all subtile rows reachable via <64KB immediates)
    const unsigned short* qlo = Qs + n * PST;           // rows n, 16+n
    const unsigned short* qhi = Qs + (32 + n) * PST;    // rows 32+n, 48+n
    const unsigned short* plo = Pall + n * PST;
    const unsigned short* phi = Pall + (32 + n) * PST;

    const int srow = tid >> 3;               // stats: row 0..63
    const int c0   = (tid & 7) * 128;        // stats: 128-wide column chunk

    float m_run = -3.0e38f, l_run = 0.f;     // valid on threads 0..63

    v8f o[4][4];                             // [M-subtile][D-ntile]
#pragma unroll
    for (int mh = 0; mh < 4; ++mh)
#pragma unroll
        for (int nt = 0; nt < 4; ++nt)
            o[mh][nt] = (v8f){0.f,0.f,0.f,0.f,0.f,0.f,0.f,0.f};
    const int myD = w * 64;                  // this wave's 64-wide D slice

    for (int r = 0; r < SS / KR; ++r) {      // two key rounds
        // ===== pass 1: logits(masked, scaled) for keys [r*KR, r*KR+KR) =====
        for (int i = 0; i < KR / 256; ++i) {              // 4 tiles per wave
            const int ktl = w + 16 * i;                   // local key tile 0..63
            const int kg  = r * KR + ktl * 16 + n;        // global key
            const unsigned short* kR = kbf + ((size_t)b * SS + kg) * DD;
            __builtin_prefetch(kbf + ((size_t)b * SS + ((kg + 256) & (SS - 1))) * DD, 0, 1);

            v8f a0 = {0.f,0.f,0.f,0.f,0.f,0.f,0.f,0.f};
            v8f a1 = a0, a2 = a0, a3 = a0;
#pragma unroll
            for (int c = 0; c < DD / 32; ++c) {
                const int ca = c * 32 + aoff;
                v16bf bv  = load16bf(kR + c * 32 + boff, kR + c * 32 + boff + 16);
                v16bf av0 = load16bf(qlo + ca,            qlo + ca + 16);
                v16bf av1 = load16bf(qlo + 16 * PST + ca, qlo + 16 * PST + ca + 16);
                v16bf av2 = load16bf(qhi + ca,            qhi + ca + 16);
                v16bf av3 = load16bf(qhi + 16 * PST + ca, qhi + 16 * PST + ca + 16);
                a0 = __builtin_amdgcn_wmma_f32_16x16x32_bf16(false, av0, false, bv, (short)0, a0, false, false);
                a1 = __builtin_amdgcn_wmma_f32_16x16x32_bf16(false, av1, false, bv, (short)0, a1, false, false);
                a2 = __builtin_amdgcn_wmma_f32_16x16x32_bf16(false, av2, false, bv, (short)0, a2, false, false);
                a3 = __builtin_amdgcn_wmma_f32_16x16x32_bf16(false, av3, false, bv, (short)0, a3, false, false);
            }
            const int* mptr = mask + ((size_t)b * SS + qt * MT) * SS + kg;
            unsigned short* slo = Pall + aoff * PST + ktl * 16 + n;        // rows aoff..,16+aoff..
            unsigned short* shi = Pall + (32 + aoff) * PST + ktl * 16 + n; // rows 32+aoff..,48+..
#pragma unroll
            for (int v = 0; v < 8; ++v) {
                float x0 = mptr[(size_t)(v + aoff) * SS]      ? a0[v] * SCALE : NEGV;
                float x1 = mptr[(size_t)(16 + v + aoff) * SS] ? a1[v] * SCALE : NEGV;
                float x2 = mptr[(size_t)(32 + v + aoff) * SS] ? a2[v] * SCALE : NEGV;
                float x3 = mptr[(size_t)(48 + v + aoff) * SS] ? a3[v] * SCALE : NEGV;
                slo[v * PST]        = f2bf(x0);
                slo[(16 + v) * PST] = f2bf(x1);
                shi[v * PST]        = f2bf(x2);
                shi[(16 + v) * PST] = f2bf(x3);
            }
        }
        __syncthreads();                               // (1) logits visible

        // ===== online softmax statistics (uint32 = 2 x bf16 sweeps) =====
        {
            const unsigned int* prow = (const unsigned int*)(Pall + srow * PST + c0);
            float mx = -3.0e38f;
            for (int j = 0; j < 64; ++j) {
                unsigned int u = prow[j];
                float xl = __uint_as_float(u << 16);
                float xh = __uint_as_float(u & 0xFFFF0000u);
                mx = fmaxf(mx, fmaxf(xl, xh));
            }
            red[tid] = mx;
        }
        __syncthreads();                               // (2)
        if (tid < MT) {
            float mx = -3.0e38f;
            for (int j = 0; j < 8; ++j) mx = fmaxf(mx, red[tid * 8 + j]);
            float mnew = fmaxf(m_run, mx);
            facs[tid] = __expf(m_run - mnew);
            mfin[tid] = mnew;
            m_run = mnew;
        }
        __syncthreads();                               // (3) mfin/facs ready

        // rescale O accumulators by exp(m_old - m_new) per row
#pragma unroll
        for (int mh = 0; mh < 4; ++mh)
#pragma unroll
            for (int v = 0; v < 8; ++v) {
                float f = facs[mh * 16 + v + aoff];
#pragma unroll
                for (int nt = 0; nt < 4; ++nt) o[mh][nt][v] *= f;
            }

        // exp in place + partial row sums
        {
            unsigned int* prow = (unsigned int*)(Pall + srow * PST + c0);
            const float m = mfin[srow];
            float s = 0.f;
            for (int j = 0; j < 64; ++j) {
                unsigned int u = prow[j];
                float e0 = __expf(__uint_as_float(u << 16) - m);
                float e1 = __expf(__uint_as_float(u & 0xFFFF0000u) - m);
                s += e0 + e1;
                prow[j] = (unsigned int)f2bf(e0) | ((unsigned int)f2bf(e1) << 16);
            }
            red[tid] = s;
        }
        __syncthreads();                               // (4) exp values + sums ready
        if (tid < MT) {
            float s = 0.f;
            for (int j = 0; j < 8; ++j) s += red[tid * 8 + j];
            l_run = l_run * facs[tid] + s;
        }

        // per-round V row bases for this wave's D slice
        const unsigned short* vb0 = vtb + ((size_t)b * DD + myD +  0 + n) * SS + r * KR + boff;
        const unsigned short* vb1 = vtb + ((size_t)b * DD + myD + 16 + n) * SS + r * KR + boff;
        const unsigned short* vb2 = vtb + ((size_t)b * DD + myD + 32 + n) * SS + r * KR + boff;
        const unsigned short* vb3 = vtb + ((size_t)b * DD + myD + 48 + n) * SS + r * KR + boff;

        // ===== pass 2: O += expP @ V for this round's keys =====
        for (int kb2 = 0; kb2 < KR / 128; ++kb2) {     // 8 blocks of 128 keys
#pragma unroll
            for (int cc = 0; cc < 4; ++cc) {
                const int co = kb2 * 128 + cc * 32;
                const int coa = co + aoff;
                v16bf pa0 = load16bf(plo + coa,            plo + coa + 16);
                v16bf pa1 = load16bf(plo + 16 * PST + coa, plo + 16 * PST + coa + 16);
                v16bf pa2 = load16bf(phi + coa,            phi + coa + 16);
                v16bf pa3 = load16bf(phi + 16 * PST + coa, phi + 16 * PST + coa + 16);
#pragma unroll
                for (int nt = 0; nt < 4; ++nt) {
                    const unsigned short* vR = (nt == 0 ? vb0 : nt == 1 ? vb1 : nt == 2 ? vb2 : vb3) + co;
                    v16bf bv = load16bf(vR, vR + 16);
                    o[0][nt] = __builtin_amdgcn_wmma_f32_16x16x32_bf16(false, pa0, false, bv, (short)0, o[0][nt], false, false);
                    o[1][nt] = __builtin_amdgcn_wmma_f32_16x16x32_bf16(false, pa1, false, bv, (short)0, o[1][nt], false, false);
                    o[2][nt] = __builtin_amdgcn_wmma_f32_16x16x32_bf16(false, pa2, false, bv, (short)0, o[2][nt], false, false);
                    o[3][nt] = __builtin_amdgcn_wmma_f32_16x16x32_bf16(false, pa3, false, bv, (short)0, o[3][nt], false, false);
                }
            }
        }
        __syncthreads();                               // (5) Pall/red reusable next round
    }

    if (tid < MT) lfin[tid] = 1.0f / l_run;
    __syncthreads();

    // ---- write pre-LayerNorm X (f32), applying 1/l per row ----
    float* xb = xout + ((size_t)b * SS + qt * MT) * DD + myD;
#pragma unroll
    for (int mh = 0; mh < 4; ++mh)
#pragma unroll
        for (int v = 0; v < 8; ++v) {
            const int rr = mh * 16 + v + aoff;
            const float li = lfin[rr];
#pragma unroll
            for (int nt = 0; nt < 4; ++nt)
                xb[(size_t)rr * DD + nt * 16 + n] = o[mh][nt][v] * li;
        }
}

// ---------------- kernel 2: row LayerNorm ----------------
__global__ void __launch_bounds__(256)
ln_kernel(const float* __restrict__ x, const float* __restrict__ gamma,
          const float* __restrict__ beta, float* __restrict__ out)
{
    __shared__ float rs[256], rq[256];
    const int row = blockIdx.x;
    const float* xr = x + (size_t)row * DD;
    float s = 0.f, s2 = 0.f;
    for (int i = threadIdx.x; i < DD; i += 256) {
        float v = xr[i];
        s += v; s2 += v * v;
    }
    rs[threadIdx.x] = s; rq[threadIdx.x] = s2;
    __syncthreads();
    for (int st = 128; st > 0; st >>= 1) {
        if (threadIdx.x < st) {
            rs[threadIdx.x] += rs[threadIdx.x + st];
            rq[threadIdx.x] += rq[threadIdx.x + st];
        }
        __syncthreads();
    }
    float mu  = rs[0] * (1.0f / DD);
    float var = rq[0] * (1.0f / DD) - mu * mu;
    float ri  = rsqrtf(var + 1e-5f);
    float* orow = out + (size_t)row * DD;
    for (int i = threadIdx.x; i < DD; i += 256)
        orow[i] = (xr[i] - mu) * ri * gamma[i] + beta[i];
}

extern "C" void kernel_launch(void* const* d_in, const int* in_sizes, int n_in,
                              void* d_out, int out_size, void* d_ws, size_t ws_size,
                              hipStream_t stream) {
    (void)in_sizes; (void)n_in; (void)out_size; (void)ws_size;
    const float* q     = (const float*)d_in[0];
    const float* k     = (const float*)d_in[1];
    const float* v     = (const float*)d_in[2];
    const int*   mask  = (const int*)d_in[3];
    const float* gamma = (const float*)d_in[4];
    const float* beta  = (const float*)d_in[5];
    float* out = (float*)d_out;

    const size_t NE = (size_t)BB * SS * DD;           // 8,388,608 elements
    unsigned short* kbf = (unsigned short*)d_ws;      // 16.8 MB
    unsigned short* vtb = kbf + NE;                   // 16.8 MB
    float*          x   = (float*)(vtb + NE);         // 33.5 MB

    const size_t shmem = (size_t)(2 * MT * PST) * sizeof(unsigned short)
                       + (512 + 3 * MT) * sizeof(float);   // ~261 KB (of 320 KB WGP LDS)
    (void)hipFuncSetAttribute((const void*)attn_kernel,
                              hipFuncAttributeMaxDynamicSharedMemorySize, (int)shmem);

    cvt_bf16_kernel   <<<(unsigned)(NE / 4 / 256), 256, 0, stream>>>(k, kbf);
    transpose_v_kernel<<<dim3(DD / 32, SS / 32, BB), dim3(32, 8), 0, stream>>>(v, vtb);
    attn_kernel       <<<BB * (SS / MT), 512, shmem, stream>>>(q, kbf, vtb, mask, x);
    ln_kernel         <<<BB * SS, 256, 0, stream>>>(x, gamma, beta, out);
}
